// TSHMBlock_64106681860638
// MI455X (gfx1250) — matrix operand order repaired and checked by reading, not compiled, not run on hardware
//
#include <hip/hip_runtime.h>
#include <hip/hip_bf16.h>
#include <math.h>

// ---------------- problem constants ----------------
#define D_MODEL 1024
#define R_DIM   32
#define K_DIM   8
#define FF_DIM  4096
#define B_SZ    4
#define L_SEQ   4096
#define M_TOK   (B_SZ * L_SEQ)          // 16384 token rows
#define CONST_W 0.001f                  // 0.1 * 0.1^2
#define EPS_G   1e-6f
#define LN_EPS  1e-5f
#define CH      256                     // scan chunk length
#define NCH     (L_SEQ / CH)            // 16 chunks per batch

// GEMM tiling
#define KSTAGE  64                      // k per LDS stage (2 WMMA substeps)
#define SBS     72                      // LDS row stride in bf16 (64 used + 8 pad)
#define SBUF_E  (128 * SBS)             // elements per LDS buffer

// ---------------- WMMA types (CDNA5 / gfx1250, wave32) ----------------
typedef __attribute__((ext_vector_type(16))) __bf16 v16bf;
typedef __attribute__((ext_vector_type(8)))  float  v8f;

union Frag16 { uint4 q[2]; v16bf v; };      // 16 bf16 per lane (8 VGPRs)
union FragC  { v8f v; float f[8]; };        // 8 f32 per lane  (8 VGPRs)

static __device__ __forceinline__ unsigned short f2bf(float f) {
    unsigned int u = __float_as_uint(f);
    unsigned int r = u + 0x7FFFu + ((u >> 16) & 1u);   // round-to-nearest-even
    return (unsigned short)(r >> 16);
}

static __device__ __forceinline__ float wave_sum(float v) {
#pragma unroll
    for (int off = 16; off > 0; off >>= 1) v += __shfl_xor(v, off, 32);
    return v;
}

// CDNA5 async global->LDS copy (16 bytes per lane), tracked by ASYNCcnt
static __device__ __forceinline__ void async_copy16(const void* gptr, void* lptr) {
    unsigned lds = (unsigned)(size_t)lptr;               // low 32 bits = LDS offset
    unsigned long long ga = (unsigned long long)(size_t)gptr;
    asm volatile("global_load_async_to_lds_b128 %0, %1, off"
                 :: "v"(lds), "v"(ga) : "memory");
}
static __device__ __forceinline__ void wait_async0() {
    asm volatile("s_wait_asynccnt 0" ::: "memory");
}

// =====================================================================
// Kernel 1: LN1 + tanh -> phi ; P = phi@Uw^T ; Q = phi@Vw^T ;
//           gate partial dots gp[t] = <X_norm, gate_w[:,t]>
// =====================================================================
__global__ __launch_bounds__(256)
void k_ln1_phi(const float* __restrict__ X,
               const float* __restrict__ Uw, const float* __restrict__ Vw,
               const float* __restrict__ gate_w,
               const float* __restrict__ ln1_g, const float* __restrict__ ln1_b,
               float* __restrict__ P, float* __restrict__ Q,
               float* __restrict__ GP)
{
    __shared__ float s_phi[D_MODEL];
    __shared__ float sred[5 * 8];
    const int row  = blockIdx.x;
    const int tid  = threadIdx.x;
    const int lane = tid & 31, wv = tid >> 5;

    const float4 x4 = ((const float4*)(X + (size_t)row * D_MODEL))[tid];
    float xv[4] = {x4.x, x4.y, x4.z, x4.w};
    float s = 0.f, ss = 0.f;
#pragma unroll
    for (int i = 0; i < 4; i++) { s += xv[i]; ss += xv[i] * xv[i]; }
    s = wave_sum(s); ss = wave_sum(ss);
    if (lane == 0) { sred[0 * 8 + wv] = s; sred[1 * 8 + wv] = ss; }
    __syncthreads();
    float mu = 0.f, m2 = 0.f;
#pragma unroll
    for (int i = 0; i < 8; i++) { mu += sred[0 * 8 + i]; m2 += sred[1 * 8 + i]; }
    mu *= (1.0f / D_MODEL);
    const float var  = m2 * (1.0f / D_MODEL) - mu * mu;
    const float rstd = rsqrtf(var + LN_EPS);

    float gp0 = 0.f, gp1 = 0.f, gp2 = 0.f;
#pragma unroll
    for (int i = 0; i < 4; i++) {
        const int d = tid * 4 + i;
        const float xn = (xv[i] - mu) * rstd * ln1_g[d] + ln1_b[d];
        s_phi[d] = tanhf(xn);
        gp0 += xn * gate_w[d * 3 + 0];
        gp1 += xn * gate_w[d * 3 + 1];
        gp2 += xn * gate_w[d * 3 + 2];
    }
    gp0 = wave_sum(gp0); gp1 = wave_sum(gp1); gp2 = wave_sum(gp2);
    if (lane == 0) { sred[2 * 8 + wv] = gp0; sred[3 * 8 + wv] = gp1; sred[4 * 8 + wv] = gp2; }
    __syncthreads();
    if (tid == 0) {
        float a = 0.f, b = 0.f, c = 0.f;
#pragma unroll
        for (int i = 0; i < 8; i++) { a += sred[16 + i]; b += sred[24 + i]; c += sred[32 + i]; }
        GP[row * 4 + 0] = a; GP[row * 4 + 1] = b; GP[row * 4 + 2] = c; GP[row * 4 + 3] = 0.f;
    }
#pragma unroll
    for (int t = 0; t < 8; t++) {
        const int j = wv * 8 + t;
        const float* w = (j < 32) ? (Uw + (size_t)j * D_MODEL)
                                  : (Vw + (size_t)(j - 32) * D_MODEL);
        float acc = 0.f;
        for (int d = lane; d < D_MODEL; d += 32) acc += s_phi[d] * w[d];
        acc = wave_sum(acc);
        if (lane == 0) {
            if (j < 32) P[(size_t)row * R_DIM + j] = acc;
            else        Q[(size_t)row * R_DIM + (j - 32)] = acc;
        }
    }
}

// =====================================================================
// Chunked forward scan of P over L, then the K=8 algebra producing M, N
// =====================================================================
__global__ __launch_bounds__(32)
void k_chunksum(const float* __restrict__ P, float* __restrict__ CP)
{
    const int b = blockIdx.x / NCH, ch = blockIdx.x % NCH, r = threadIdx.x;
    const size_t base = ((size_t)b * L_SEQ + ch * CH) * R_DIM + r;
    float s = 0.f;
    for (int l = 0; l < CH; l++) s += P[base + (size_t)l * R_DIM];
    CP[(size_t)blockIdx.x * R_DIM + r] = s;
}

__global__ __launch_bounds__(32)
void k_chunkpref(float* __restrict__ CP)   // exclusive prefix over chunks
{
    const int b = blockIdx.x, r = threadIdx.x;
    float run = 0.f;
    for (int ch = 0; ch < NCH; ch++) {
        const size_t i = (size_t)(b * NCH + ch) * R_DIM + r;
        const float t = CP[i]; CP[i] = run; run += t;
    }
}

__global__ __launch_bounds__(32)
void k_scan_main(const float* __restrict__ P, const float* __restrict__ Q,
                 const float* __restrict__ CP,
                 const float* __restrict__ Aw, const float* __restrict__ Ab,
                 const float* __restrict__ Bw, const float* __restrict__ Bb,
                 const float* __restrict__ cc,
                 float* __restrict__ Mb, float* __restrict__ Nb,
                 float* __restrict__ CM)
{
    const int b = blockIdx.x / NCH, ch = blockIdx.x % NCH, r = threadIdx.x;
    float awr[K_DIM], bwr[K_DIM], base[K_DIM];
#pragma unroll
    for (int k = 0; k < K_DIM; k++) {
        awr[k] = Aw[k * R_DIM + r]; bwr[k] = Bw[k * R_DIM + r];
        base[k] = Ab[k] + Bb[k] + cc[k];
    }
    float S = CP[(size_t)blockIdx.x * R_DIM + r];
    float msum = 0.f;
    const size_t rowbase = ((size_t)b * L_SEQ + ch * CH) * R_DIM + r;
    for (int l = 0; l < CH; l++) {
        const size_t idx = rowbase + (size_t)l * R_DIM;
        S += P[idx];
        const float q = Q[idx];
        float m = 0.f, n = 0.f;
#pragma unroll
        for (int k = 0; k < K_DIM; k++) {
            const float t = wave_sum(S * awr[k] + q * bwr[k]);   // all lanes get sum
            const float G = CONST_W * tanhf(t + base[k]);
            m += G * awr[k]; n += G * bwr[k];
        }
        Mb[idx] = m; Nb[idx] = n; msum += m;
    }
    CM[(size_t)blockIdx.x * R_DIM + r] = msum;
}

__global__ __launch_bounds__(32)
void k_chunksuf(float* __restrict__ CM)    // exclusive suffix over chunks
{
    const int b = blockIdx.x, r = threadIdx.x;
    float run = 0.f;
    for (int ch = NCH - 1; ch >= 0; ch--) {
        const size_t i = (size_t)(b * NCH + ch) * R_DIM + r;
        const float t = CM[i]; CM[i] = run; run += t;
    }
}

__global__ __launch_bounds__(32)
void k_suffix_main(const float* __restrict__ Mb, const float* __restrict__ CM,
                   float* __restrict__ Msuf)
{
    const int b = blockIdx.x / NCH, ch = blockIdx.x % NCH, r = threadIdx.x;
    float Ms = CM[(size_t)blockIdx.x * R_DIM + r];
    const size_t rowbase = ((size_t)b * L_SEQ + ch * CH) * R_DIM + r;
    for (int l = CH - 1; l >= 0; l--) {
        const size_t idx = rowbase + (size_t)l * R_DIM;
        Ms += Mb[idx]; Msuf[idx] = Ms;
    }
}

// =====================================================================
// grad = Msuf@Uw + N@Vw ; gate conv ; residual ; LN2 -> bf16 H
// =====================================================================
__global__ __launch_bounds__(256)
void k_grad_res_ln2(const float* __restrict__ X,
                    const float* __restrict__ Uw, const float* __restrict__ Vw,
                    const float* __restrict__ Msuf, const float* __restrict__ Nb,
                    const float* __restrict__ GP,
                    const float* __restrict__ gate_b, const float* __restrict__ res_scale,
                    const float* __restrict__ layer_scale,
                    const float* __restrict__ ln2_g, const float* __restrict__ ln2_b,
                    float* __restrict__ Xnext, unsigned short* __restrict__ Hbf)
{
    __shared__ float s_m[R_DIM], s_n[R_DIM];
    __shared__ float sred[2 * 8];
    const int row = blockIdx.x;
    const int l   = row % L_SEQ;
    const int tid = threadIdx.x, lane = tid & 31, wv = tid >> 5;
    if (tid < R_DIM) {
        s_m[tid] = Msuf[(size_t)row * R_DIM + tid];
        s_n[tid] = Nb[(size_t)row * R_DIM + tid];
    }
    __syncthreads();
    float graw = gate_b[0] + GP[row * 4 + 1];
    if (l > 0)          graw += GP[(row - 1) * 4 + 0];
    if (l < L_SEQ - 1)  graw += GP[(row + 1) * 4 + 2];
    float g = 1.0f / (1.0f + expf(-graw));
    g = EPS_G + (1.0f - EPS_G) * g;
    const float scl = res_scale[0] * g;

    float xv[4]; float s = 0.f, ss = 0.f;
#pragma unroll
    for (int i = 0; i < 4; i++) {
        const int d = tid * 4 + i;
        float grad = 0.f;
#pragma unroll
        for (int r = 0; r < R_DIM; r++)
            grad += s_m[r] * Uw[r * D_MODEL + d] + s_n[r] * Vw[r * D_MODEL + d];
        const float v = X[(size_t)row * D_MODEL + d] + scl * grad * layer_scale[d];
        Xnext[(size_t)row * D_MODEL + d] = v;
        xv[i] = v; s += v; ss += v * v;
    }
    s = wave_sum(s); ss = wave_sum(ss);
    if (lane == 0) { sred[wv] = s; sred[8 + wv] = ss; }
    __syncthreads();
    float mu = 0.f, m2 = 0.f;
#pragma unroll
    for (int i = 0; i < 8; i++) { mu += sred[i]; m2 += sred[8 + i]; }
    mu *= (1.0f / D_MODEL);
    const float rstd = rsqrtf(m2 * (1.0f / D_MODEL) - mu * mu + LN_EPS);
#pragma unroll
    for (int i = 0; i < 4; i++) {
        const int d = tid * 4 + i;
        const float h = (xv[i] - mu) * rstd * ln2_g[d] + ln2_b[d];
        Hbf[(size_t)row * D_MODEL + d] = f2bf(h);
    }
}

// =====================================================================
// f32 -> bf16 weight conversion (W1 and W2, same element count)
// =====================================================================
__global__ __launch_bounds__(256)
void k_cvt(const float* __restrict__ a, const float* __restrict__ b,
           unsigned short* __restrict__ oa, unsigned short* __restrict__ ob, int n)
{
    for (int i = blockIdx.x * blockDim.x + threadIdx.x; i < n; i += gridDim.x * blockDim.x) {
        oa[i] = f2bf(a[i]); ob[i] = f2bf(b[i]);
    }
}

// =====================================================================
// WMMA GEMM core: 128x128 tile per block (8 waves x 16x128 strips).
//  - B tile (128 n x 64 k bf16) staged in double-buffered LDS via
//    global_load_async_to_lds_b128 (ASYNCcnt), overlapped with compute.
//  - Per 32-k substep: all 8 B frags loaded (kept live via sched_barrier),
//    then 8 back-to-back WMMAs with a single s_wait_dscnt.
// =====================================================================
template<int KDIM>
static __device__ __forceinline__ void wmma_gemm_core(
    const unsigned short* __restrict__ A,   // [M, KDIM] bf16
    const unsigned short* __restrict__ Bw,  // [N, KDIM] bf16 (contraction contiguous)
    unsigned short* sB,                     // LDS: 2 * SBUF_E
    int m0_block, int n0, int tid, FragC* acc)
{
    const int lane = tid & 31, wave = tid >> 5;
    const int h = lane >> 4, mn = lane & 15;
    const int m0 = m0_block + wave * 16;
    const unsigned short* arow = A + (size_t)(m0 + mn) * KDIM;

    // async-copy one 64-k stage of B into LDS buffer `buf`
    // 128 rows x 8 chunks(16B) = 1024 copies; 256 threads x 4
    auto issue_stage = [&](int kk, int buf) {
#pragma unroll
        for (int i = 0; i < 4; i++) {
            const int idx  = tid + 256 * i;
            const int row  = idx >> 3;        // 0..127
            const int chk  = idx & 7;         // 0..7 (8 bf16 each)
            const unsigned short* src = Bw + (size_t)(n0 + row) * KDIM + kk + chk * 8;
            unsigned short* dst = sB + buf * SBUF_E + row * SBS + chk * 8;
            async_copy16(src, dst);
        }
    };

    issue_stage(0, 0);
    wait_async0();
    __syncthreads();

    int cur = 0;
    for (int kk = 0; kk < KDIM; kk += KSTAGE) {
        if (kk + KSTAGE < KDIM) issue_stage(kk + KSTAGE, cur ^ 1);

        const unsigned short* sbuf = sB + cur * SBUF_E;
#pragma unroll
        for (int ks = 0; ks < KSTAGE; ks += 32) {
            Frag16 a;
            a.q[0] = *(const uint4*)(arow + kk + ks + h * 16);
            a.q[1] = *(const uint4*)(arow + kk + ks + h * 16 + 8);
            Frag16 bf[8];
#pragma unroll
            for (int j = 0; j < 8; j++) {
                const unsigned short* p = sbuf + (j * 16 + mn) * SBS + ks + h * 16;
                bf[j].q[0] = *(const uint4*)(p);
                bf[j].q[1] = *(const uint4*)(p + 8);
            }
            // keep all 8 B fragments live: loads may not sink past this point,
            // WMMAs may not hoist above it -> one dscnt wait, 8 chained WMMAs
            __builtin_amdgcn_sched_barrier(0);
#pragma unroll
            for (int j = 0; j < 8; j++)
                acc[j].v = __builtin_amdgcn_wmma_f32_16x16x32_bf16(
                    false, a.v, false, bf[j].v, (short)0, acc[j].v, false, false);
        }
        wait_async0();        // next-stage copies done (issued before compute)
        __syncthreads();      // everyone finished reading `cur` before overwrite
        cur ^= 1;
    }
}

// GEMM 1: Act = gelu(H[M,1024] @ W1^T + b1) -> bf16 [M, FF]
__global__ __launch_bounds__(256)
void k_ffn1(const unsigned short* __restrict__ Hbf,
            const unsigned short* __restrict__ W1bf,
            const float* __restrict__ b1,
            unsigned short* __restrict__ Act)
{
    __shared__ unsigned short sB[2 * SBUF_E];
    const int tid = threadIdx.x;
    const int lane = tid & 31, wave = tid >> 5;
    const int h = lane >> 4, mn = lane & 15;
    const int m0b = blockIdx.y * 128, n0 = blockIdx.x * 128;

    FragC acc[8];
#pragma unroll
    for (int j = 0; j < 8; j++)
#pragma unroll
        for (int i = 0; i < 8; i++) acc[j].f[i] = 0.f;

    wmma_gemm_core<D_MODEL>(Hbf, W1bf, sB, m0b, n0, tid, acc);

#pragma unroll
    for (int j = 0; j < 8; j++) {
        const int n = n0 + j * 16 + mn;
        const float bias = b1[n];
#pragma unroll
        for (int i = 0; i < 8; i++) {
            const int m = m0b + wave * 16 + h * 8 + i;
            const float x = acc[j].f[i] + bias;
            const float y = 0.5f * x * (1.0f + erff(x * 0.70710678118f));  // exact gelu
            Act[(size_t)m * FF_DIM + n] = f2bf(y);
        }
    }
}

// GEMM 2: out = Xnext + Act[M,FF] @ W2^T + b2 -> f32 [M, 1024]
__global__ __launch_bounds__(256)
void k_ffn2(const unsigned short* __restrict__ Act,
            const unsigned short* __restrict__ W2bf,
            const float* __restrict__ b2,
            const float* __restrict__ Xnext,
            float* __restrict__ Out)
{
    __shared__ unsigned short sB[2 * SBUF_E];
    const int tid = threadIdx.x;
    const int lane = tid & 31, wave = tid >> 5;
    const int h = lane >> 4, mn = lane & 15;
    const int m0b = blockIdx.y * 128, n0 = blockIdx.x * 128;

    FragC acc[8];
#pragma unroll
    for (int j = 0; j < 8; j++)
#pragma unroll
        for (int i = 0; i < 8; i++) acc[j].f[i] = 0.f;

    wmma_gemm_core<FF_DIM>(Act, W2bf, sB, m0b, n0, tid, acc);

#pragma unroll
    for (int j = 0; j < 8; j++) {
        const int n = n0 + j * 16 + mn;
        const float bias = b2[n];
#pragma unroll
        for (int i = 0; i < 8; i++) {
            const int m = m0b + wave * 16 + h * 8 + i;
            const size_t o = (size_t)m * D_MODEL + n;
            Out[o] = Xnext[o] + acc[j].f[i] + bias;
        }
    }
}

// =====================================================================
extern "C" void kernel_launch(void* const* d_in, const int* in_sizes, int n_in,
                              void* d_out, int out_size, void* d_ws, size_t ws_size,
                              hipStream_t stream)
{
    const float* X       = (const float*)d_in[0];
    const float* Uw      = (const float*)d_in[1];
    const float* Vw      = (const float*)d_in[2];
    const float* Aw      = (const float*)d_in[3];
    const float* Ab      = (const float*)d_in[4];
    const float* Bw      = (const float*)d_in[5];
    const float* Bb      = (const float*)d_in[6];
    const float* cvec    = (const float*)d_in[7];
    const float* gate_w  = (const float*)d_in[8];
    const float* gate_b  = (const float*)d_in[9];
    const float* res_sc  = (const float*)d_in[10];
    const float* lsc     = (const float*)d_in[11];
    const float* ln1_g   = (const float*)d_in[12];
    const float* ln1_b   = (const float*)d_in[13];
    const float* ln2_g   = (const float*)d_in[14];
    const float* ln2_b   = (const float*)d_in[15];
    const float* W1      = (const float*)d_in[16];
    const float* b1      = (const float*)d_in[17];
    const float* W2      = (const float*)d_in[18];
    const float* b2      = (const float*)d_in[19];
    float* Out = (float*)d_out;

    char* ws = (char*)d_ws;
    size_t off = 0;
    auto carve = [&](size_t bytes) -> void* {
        void* p = ws + off; off += (bytes + 255) & ~(size_t)255; return p;
    };
    float* P    = (float*)carve((size_t)M_TOK * R_DIM * 4);
    float* Q    = (float*)carve((size_t)M_TOK * R_DIM * 4);
    float* GP   = (float*)carve((size_t)M_TOK * 4 * 4);
    float* Mb   = (float*)carve((size_t)M_TOK * R_DIM * 4);
    float* Nb   = (float*)carve((size_t)M_TOK * R_DIM * 4);
    float* Msuf = (float*)carve((size_t)M_TOK * R_DIM * 4);
    float* CP   = (float*)carve((size_t)B_SZ * NCH * R_DIM * 4);
    float* CM   = (float*)carve((size_t)B_SZ * NCH * R_DIM * 4);
    float* Xnx  = (float*)carve((size_t)M_TOK * D_MODEL * 4);
    unsigned short* Hbf  = (unsigned short*)carve((size_t)M_TOK * D_MODEL * 2);
    unsigned short* W1bf = (unsigned short*)carve((size_t)FF_DIM * D_MODEL * 2);
    unsigned short* W2bf = (unsigned short*)carve((size_t)D_MODEL * FF_DIM * 2);
    unsigned short* Act  = (unsigned short*)carve((size_t)M_TOK * FF_DIM * 2);

    k_cvt<<<4096, 256, 0, stream>>>(W1, W2, W1bf, W2bf, FF_DIM * D_MODEL);

    k_ln1_phi<<<M_TOK, 256, 0, stream>>>(X, Uw, Vw, gate_w, ln1_g, ln1_b, P, Q, GP);

    k_chunksum<<<B_SZ * NCH, 32, 0, stream>>>(P, CP);
    k_chunkpref<<<B_SZ, 32, 0, stream>>>(CP);
    k_scan_main<<<B_SZ * NCH, 32, 0, stream>>>(P, Q, CP, Aw, Ab, Bw, Bb, cvec, Mb, Nb, CM);
    k_chunksuf<<<B_SZ, 32, 0, stream>>>(CM);
    k_suffix_main<<<B_SZ * NCH, 32, 0, stream>>>(Mb, CM, Msuf);

    k_grad_res_ln2<<<M_TOK, 256, 0, stream>>>(X, Uw, Vw, Msuf, Nb, GP, gate_b, res_sc,
                                              lsc, ln2_g, ln2_b, Xnx, Hbf);

    dim3 g1(FF_DIM / 128, M_TOK / 128);
    k_ffn1<<<g1, 256, 0, stream>>>(Hbf, W1bf, b1, Act);

    dim3 g2(D_MODEL / 128, M_TOK / 128);
    k_ffn2<<<g2, 256, 0, stream>>>(Act, W2bf, b2, Xnx, Out);
}
